// NeighborAttention_82970178224287
// MI455X (gfx1250) — compile-verified, matrix operand order checked by reading, not run on hardware
//
#include <hip/hip_runtime.h>

// Problem constants (B=4, N=2000, K=30, H=128, HEADS=4, D=32, NUM_IN=256)
#define B_     4
#define N_     2000
#define KNB    30
#define H_     128
#define HEADS_ 4
#define D_     32
#define NUMIN  256
#define BN     (B_ * N_)          // 8000

typedef __attribute__((ext_vector_type(16))) _Float16     v16h;
typedef __attribute__((ext_vector_type(8)))  _Float16     v8h;
typedef __attribute__((ext_vector_type(8)))  float        v8f;
typedef __attribute__((ext_vector_type(4)))  unsigned int v4u;
typedef __attribute__((ext_vector_type(8)))  int          v8i;
typedef __attribute__((ext_vector_type(4)))  int          v4i;

// ---------------------------------------------------------------------------
// WMMA helpers (gfx1250: V_WMMA_F32_16X16X32_F16, wave32)
// A 16x32 f16: lane&15 = row; lanes 0-15 hold K {k0..k0+7, k0+16..k0+23},
// lanes 16-31 hold K {k0+8..k0+15, k0+24..k0+31} -> two 16B chunks per lane.
// B (32x16 = K x N): same pattern with "row" = output column n; for torch
// Linear (y = x @ W^T), B[k][n] = W[n][k], i.e. rows of row-major W directly.
// ---------------------------------------------------------------------------
__device__ __forceinline__ v16h load_frag16(const _Float16* __restrict__ base,
                                            int row0, int col0, int ld, int lane) {
  const int r  = lane & 15;
  const int kb = (lane & 16) ? 8 : 0;
  const _Float16* p = base + (row0 + r) * ld + col0 + kb;
  v8h lo = *(const v8h*)(p);
  v8h hi = *(const v8h*)(p + 16);
  return __builtin_shufflevector(lo, hi, 0, 1, 2, 3, 4, 5, 6, 7,
                                 8, 9, 10, 11, 12, 13, 14, 15);
}

__device__ __forceinline__ v8f wmma16(v16h a, v16h b, v8f c) {
  return __builtin_amdgcn_wmma_f32_16x16x32_f16(false, a, false, b,
                                                (short)0, c, false, false);
}

// C/D 16x16 f32 layout: lane&15 = col; VGPR r = row r (lanes 0-15) / r+8.
__device__ __forceinline__ void store_frag_f32(float* __restrict__ dst,
                                               int row0, int col0, int ld,
                                               int lane, v8f d) {
  const int r0 = row0 + ((lane & 16) ? 8 : 0);
  const int c  = col0 + (lane & 15);
#pragma unroll
  for (int r = 0; r < 8; ++r) dst[(r0 + r) * ld + c] = d[r];
}

__device__ __forceinline__ void store_frag_f16(_Float16* __restrict__ dst,
                                               int row0, int col0, int ld,
                                               int lane, v8f d) {
  const int r0 = row0 + ((lane & 16) ? 8 : 0);
  const int c  = col0 + (lane & 15);
#pragma unroll
  for (int r = 0; r < 8; ++r) dst[(r0 + r) * ld + c] = (_Float16)d[r];
}

// 32x128 GEMM slab with f16 result: Dst(32x128) = A(32 x 32*ktiles) @ W^T.
// 4 waves; wave w owns output columns [32w, 32w+32).
__device__ __forceinline__ void gemm32x128_h(const _Float16* __restrict__ A, int lda,
                                             const _Float16* __restrict__ W, int ldb,
                                             _Float16* __restrict__ Dst,
                                             int ktiles, int wave, int lane) {
  v8f acc00 = {}, acc01 = {}, acc10 = {}, acc11 = {};
  for (int kt = 0; kt < ktiles; ++kt) {
    const int k0 = kt * 32;
    v16h a0 = load_frag16(A, 0,  k0, lda, lane);
    v16h a1 = load_frag16(A, 16, k0, lda, lane);
    v16h b0 = load_frag16(W, (wave * 2 + 0) * 16, k0, ldb, lane);
    v16h b1 = load_frag16(W, (wave * 2 + 1) * 16, k0, ldb, lane);
    acc00 = wmma16(a0, b0, acc00);
    acc01 = wmma16(a0, b1, acc01);
    acc10 = wmma16(a1, b0, acc10);
    acc11 = wmma16(a1, b1, acc11);
  }
  store_frag_f16(Dst, 0,  (wave * 2 + 0) * 16, H_, lane, acc00);
  store_frag_f16(Dst, 0,  (wave * 2 + 1) * 16, H_, lane, acc01);
  store_frag_f16(Dst, 16, (wave * 2 + 0) * 16, H_, lane, acc10);
  store_frag_f16(Dst, 16, (wave * 2 + 1) * 16, H_, lane, acc11);
}

// ---------------------------------------------------------------------------
// Tensor Data Mover: DMA a 2D f32 tile (rows x width, contiguous rows of
// `width` elements) from global memory into LDS. D# packing per ISA 8.3/8.4:
//   g0: [1:0]count=1 | [63:32]lds_addr | [120:64]global_addr | [127:126]type=2
//   g1: data_size=4B; tensor_dim0/1 = width/rows; tile_dim0/1 = width/rows;
//       tensor_dim0_stride = width. Groups 2/3 zero (2D tensor).
// Tracked by TENSORcnt. EXEC ignored; call from one wave only.
// This toolchain's builtin is the 6-arg form:
//   (uint32x4 g0, int32x8 g1, int32x4 g2, int32x4 g3, int32x8 g4, i32 cpol)
// ---------------------------------------------------------------------------
__device__ __forceinline__ void tdm_load_tile_f32(unsigned int lds_off,
                                                  const float* __restrict__ gsrc,
                                                  unsigned int width,
                                                  unsigned int rows) {
  const unsigned long long ga = (unsigned long long)(uintptr_t)gsrc;
  v4u g0;
  g0[0] = 1u;                                            // count=1, user mode
  g0[1] = lds_off;                                       // lds_addr (bytes)
  g0[2] = (unsigned int)(ga & 0xFFFFFFFFu);              // global_addr[31:0]
  g0[3] = (unsigned int)((ga >> 32) & 0x1FFFFFFu)        // global_addr[56:32]
          | (2u << 30);                                  // type = 2 (image)
  v8i g1;
  g1[0] = (int)(2u << 16);                 // data_size = 2 -> 4 bytes
  g1[1] = (int)(width << 16);              // tensor_dim0[15:0] @ bits[63:48]
  g1[2] = (int)((width >> 16) | (rows << 16)); // dim0 hi | tensor_dim1 lo
  g1[3] = (int)((rows >> 16) | (width << 16)); // dim1 hi | tile_dim0
  g1[4] = (int)rows;                       // tile_dim1 (tile_dim2 = 0)
  g1[5] = (int)width;                      // tensor_dim0_stride[31:0]
  g1[6] = 0;
  g1[7] = 0;
  v4i z4 = {0, 0, 0, 0};
  v8i z8 = {0, 0, 0, 0, 0, 0, 0, 0};
  __builtin_amdgcn_tensor_load_to_lds(g0, g1, z4, z4, z8, 0);
}

__device__ __forceinline__ unsigned int lds_byte_offset(const void* p) {
  // Flat shared-aperture address: addr[31:0] is the LDS byte offset.
  return (unsigned int)(uintptr_t)p;
}

// ---------------------------------------------------------------------------
// Kernel A: f32 -> f16 conversion (weights + h_V), done once.
// ---------------------------------------------------------------------------
__global__ __launch_bounds__(256) void cvt_f32_to_f16(const float* __restrict__ src,
                                                      _Float16* __restrict__ dst, int n) {
  for (int i = blockIdx.x * blockDim.x + threadIdx.x; i < n;
       i += gridDim.x * blockDim.x)
    dst[i] = (_Float16)src[i];
}

// ---------------------------------------------------------------------------
// Kernels B/D: Out[M][128] (f32) = A[M][128] (f16) @ W[128][128]^T (f16).
// One block = 32 rows, 4 waves, 16 WMMAs/wave. Grid = M/32.
// ---------------------------------------------------------------------------
__global__ __launch_bounds__(128) void gemm_MxH(const _Float16* __restrict__ A,
                                                const _Float16* __restrict__ W,
                                                float* __restrict__ Out) {
  const int wave = threadIdx.x >> 5, lane = threadIdx.x & 31;
  const int row0 = blockIdx.x * 32;
  const _Float16* Ab = A + (long)row0 * H_;
  v8f acc00 = {}, acc01 = {}, acc10 = {}, acc11 = {};
#pragma unroll
  for (int kt = 0; kt < 4; ++kt) {
    const int k0 = kt * 32;
    v16h a0 = load_frag16(Ab, 0,  k0, H_, lane);
    v16h a1 = load_frag16(Ab, 16, k0, H_, lane);
    v16h b0 = load_frag16(W, (wave * 2 + 0) * 16, k0, H_, lane);
    v16h b1 = load_frag16(W, (wave * 2 + 1) * 16, k0, H_, lane);
    acc00 = wmma16(a0, b0, acc00);
    acc01 = wmma16(a0, b1, acc01);
    acc10 = wmma16(a1, b0, acc10);
    acc11 = wmma16(a1, b1, acc11);
  }
  float* Ob = Out + (long)row0 * H_;
  store_frag_f32(Ob, 0,  (wave * 2 + 0) * 16, H_, lane, acc00);
  store_frag_f32(Ob, 0,  (wave * 2 + 1) * 16, H_, lane, acc01);
  store_frag_f32(Ob, 16, (wave * 2 + 0) * 16, H_, lane, acc10);
  store_frag_f32(Ob, 16, (wave * 2 + 1) * 16, H_, lane, acc11);
}

// ---------------------------------------------------------------------------
// Kernel C: fused neighbor attention core. One block per (b,n) token.
// 128 threads = 4 wave32s. LDS ~63 KB:
//   sF32  : 30x256 f32  TDM landing zone (KE+KV first, then EV, prefetched)
//   sA16  : 32x256 f16  GEMM A operand (KE16|KV16, later EV16)
//   sK1h  : 32x128 f16  K1, reused for V
//   sK2h  : 32x128 f16  K2
// ---------------------------------------------------------------------------
__global__ __launch_bounds__(128) void neigh_attn_core(
    const float* __restrict__ hEV, const float* __restrict__ hKV,
    const float* __restrict__ hKE, const float* __restrict__ maskA,
    const float* __restrict__ Qall,
    const _Float16* __restrict__ Wk1, const _Float16* __restrict__ Wk2,
    const _Float16* __restrict__ Wv,
    _Float16* __restrict__ hupd) {
  __shared__ float    sF32[KNB * NUMIN];     // 30720 B
  __shared__ _Float16 sA16[32 * NUMIN];      // 16384 B
  __shared__ _Float16 sK1h[32 * H_];         //  8192 B (later V)
  __shared__ _Float16 sK2h[32 * H_];         //  8192 B
  __shared__ float    sQ[H_];
  __shared__ float    sMask[32];
  __shared__ float    sAtt[HEADS_ * 32];

  const int bn   = blockIdx.x;
  const int tid  = threadIdx.x;
  const int wave = tid >> 5, lane = tid & 31;

  const float* keb = hKE + (long)bn * KNB * H_;
  const float* kvb = hKV + (long)bn * KNB * H_;
  const float* evb = hEV + (long)bn * KNB * NUMIN;

  // TDM: DMA KE (30x128 f32) and KV tiles into LDS; wave 0 issues.
  if (tid < 32) {
    const unsigned int base = lds_byte_offset(sF32);
    tdm_load_tile_f32(base,                keb, H_, KNB);
    tdm_load_tile_f32(base + KNB * H_ * 4, kvb, H_, KNB);
  }
  sQ[tid] = Qall[(long)bn * H_ + tid];
  if (tid < 32) sMask[tid] = (tid < KNB) ? maskA[(long)bn * KNB + tid] : 0.f;
  __builtin_amdgcn_s_wait_tensorcnt(0);
  __syncthreads();

  // Convert KE/KV f32 -> f16 (zero-pad rows 30,31).
  for (int i = tid; i < 32 * H_; i += 128) {
    const int r = i >> 7, c = i & (H_ - 1);
    const bool in = (r < KNB);
    sA16[i]        = in ? (_Float16)sF32[r * H_ + c]            : (_Float16)0.f;
    sA16[4096 + i] = in ? (_Float16)sF32[KNB * H_ + r * H_ + c] : (_Float16)0.f;
  }
  __syncthreads();

  // Prefetch EV tile via TDM; streams in under GEMM1/GEMM2/logits/softmax.
  if (tid < 32) {
    tdm_load_tile_f32(lds_byte_offset(sF32), evb, NUMIN, KNB);
  }

  // K1 = KE16 @ W_K1^T ; K2 = KV16 @ W_K2^T (f16 results in LDS)
  gemm32x128_h(sA16,        H_, Wk1, H_, sK1h, 4, wave, lane);
  gemm32x128_h(sA16 + 4096, H_, Wk2, H_, sK2h, 4, wave, lane);
  __syncthreads();

  // Trilinear logits (scaled 1/D) + mask.
  if (tid < KNB * HEADS_) {
    const int k = tid >> 2, h = tid & 3;
    const float*    q  = sQ   + h * D_;
    const _Float16* p1 = sK1h + k * H_ + h * D_;
    const _Float16* p2 = sK2h + k * H_ + h * D_;
    float acc = 0.f;
#pragma unroll
    for (int d = 0; d < D_; ++d)
      acc = fmaf(q[d] * (float)p1[d], (float)p2[d], acc);
    acc *= (1.f / (float)D_);
    sAtt[h * 32 + k] = (sMask[k] > 0.f) ? acc : -3.0e38f;
  }
  __syncthreads();

  // Per-head masked softmax over neighbors.
  if (tid < HEADS_) {
    const int h = tid;
    float m = -3.4e38f;
    for (int k = 0; k < KNB; ++k) m = fmaxf(m, sAtt[h * 32 + k]);
    float s = 0.f;
    for (int k = 0; k < KNB; ++k) s += __expf(sAtt[h * 32 + k] - m);
    const float inv = (s > 0.f) ? (1.f / s) : 0.f;
    for (int k = 0; k < KNB; ++k) {
      const float a = __expf(sAtt[h * 32 + k] - m) * inv;
      sAtt[h * 32 + k] = (sMask[k] > 0.f) ? a : 0.f;
    }
    for (int k = KNB; k < 32; ++k) sAtt[h * 32 + k] = 0.f;
  }
  __builtin_amdgcn_s_wait_tensorcnt(0);   // EV tile landed (wave 0's counter)
  __syncthreads();

  // Convert EV f32 -> f16 (32x256, zero-pad rows 30,31).
  for (int i = tid; i < 32 * NUMIN; i += 128) {
    const int r = i >> 8, c = i & (NUMIN - 1);
    sA16[i] = (r < KNB) ? (_Float16)sF32[r * NUMIN + c] : (_Float16)0.f;
  }
  __syncthreads();

  // V = EV16 @ W_V^T (f16 result reuses sK1h; K1 already consumed).
  gemm32x128_h(sA16, NUMIN, Wv, NUMIN, sK1h, 8, wave, lane);
  __syncthreads();

  // Weighted aggregation over neighbors: h_upd[h*32+d], f32 accumulate.
  {
    const int h = tid >> 5;  // tid == h*32 + d
    float acc = 0.f;
    for (int k = 0; k < KNB; ++k)
      acc = fmaf(sAtt[h * 32 + k], (float)sK1h[k * H_ + tid], acc);
    hupd[(long)bn * H_ + tid] = (_Float16)acc;
  }
}

// ---------------------------------------------------------------------------
// Host launcher. Inputs (setup_inputs order):
// 0:h_V 1:h_EV 2:h_KV 3:h_KE 4:mask_attend 5:W_Q 6:W_K1 7:W_K2 8:W_V 9:W_O
// Workspace (~8.4 MB): f16 h_V + weights, f32 Q, f16 h_upd.
// ---------------------------------------------------------------------------
extern "C" void kernel_launch(void* const* d_in, const int* in_sizes, int n_in,
                              void* d_out, int out_size, void* d_ws, size_t ws_size,
                              hipStream_t stream) {
  (void)in_sizes; (void)n_in; (void)out_size; (void)ws_size;
  const float* h_V   = (const float*)d_in[0];
  const float* h_EV  = (const float*)d_in[1];
  const float* h_KV  = (const float*)d_in[2];
  const float* h_KE  = (const float*)d_in[3];
  const float* maskA = (const float*)d_in[4];
  const float* W_Q   = (const float*)d_in[5];
  const float* W_K1  = (const float*)d_in[6];
  const float* W_K2  = (const float*)d_in[7];
  const float* W_V   = (const float*)d_in[8];
  const float* W_O   = (const float*)d_in[9];

  char* p = (char*)d_ws;
  auto take = [&](size_t bytes) {
    char* r = p;
    p += (bytes + 255) & ~(size_t)255;
    return r;
  };
  _Float16* hV16  = (_Float16*)take((size_t)BN * H_ * 2);
  _Float16* Wq16  = (_Float16*)take((size_t)H_ * H_ * 2);
  _Float16* Wk116 = (_Float16*)take((size_t)H_ * H_ * 2);
  _Float16* Wk216 = (_Float16*)take((size_t)H_ * H_ * 2);
  _Float16* Wo16  = (_Float16*)take((size_t)H_ * H_ * 2);
  _Float16* Wv16  = (_Float16*)take((size_t)H_ * NUMIN * 2);
  float*    Qall  = (float*)take((size_t)BN * H_ * 4);
  _Float16* hupd  = (_Float16*)take((size_t)BN * H_ * 2);

  cvt_f32_to_f16<<<512, 256, 0, stream>>>(h_V, hV16, BN * H_);
  cvt_f32_to_f16<<<64, 256, 0, stream>>>(W_Q, Wq16, H_ * H_);
  cvt_f32_to_f16<<<64, 256, 0, stream>>>(W_K1, Wk116, H_ * H_);
  cvt_f32_to_f16<<<64, 256, 0, stream>>>(W_K2, Wk216, H_ * H_);
  cvt_f32_to_f16<<<64, 256, 0, stream>>>(W_O, Wo16, H_ * H_);
  cvt_f32_to_f16<<<128, 256, 0, stream>>>(W_V, Wv16, H_ * NUMIN);

  // Q = h_V @ W_Q^T  (WMMA, f32 out)
  gemm_MxH<<<BN / 32, 128, 0, stream>>>(hV16, Wq16, Qall);

  // Fused neighbor attention core (one block per token, TDM-staged tiles).
  neigh_attn_core<<<BN, 128, 0, stream>>>(h_EV, h_KV, h_KE, maskA, Qall,
                                          Wk116, Wk216, Wv16, hupd);

  // out = h_upd @ W_O^T  (WMMA, f32 out)
  gemm_MxH<<<BN / 32, 128, 0, stream>>>(hupd, Wo16, (float*)d_out);
}
